// LatentODE_79276506349919
// MI455X (gfx1250) — matrix operand order repaired and checked by reading, not compiled, not run on hardware
//
#include <hip/hip_runtime.h>

// Problem constants (match reference)
#define B_TOT  1024
#define S_LEN  64
#define D_IN   32
#define L_DIM  64
#define H_DIM  512
#define T_EVAL 128
#define NSUB   4

typedef __attribute__((ext_vector_type(16))) _Float16 v16h;
typedef __attribute__((ext_vector_type(8)))  float    v8f;
typedef __attribute__((ext_vector_type(4)))  unsigned v4u;

union HFrag { v16h h; v4u q[2]; unsigned u[8]; _Float16 e[16]; };
union AccU  { v8f v; float f[8]; };
union Pack2 { _Float16 e[2]; unsigned u; };

static __device__ __forceinline__ void zero8(AccU& a){
#pragma unroll
  for (int r = 0; r < 8; ++r) a.f[r] = 0.f;
}

// ---- native transcendentals (gfx1250 has v_tanh_f32 / v_exp_f32 / v_rcp_f32) ----
static __device__ __forceinline__ float fast_tanh(float x){
#if __has_builtin(__builtin_amdgcn_tanhf)
  return __builtin_amdgcn_tanhf(x);
#else
  const float e = __expf(2.0f * x);
  return 1.0f - 2.0f / (e + 1.0f);
#endif
}
static __device__ __forceinline__ float fast_rcp(float x){
#if __has_builtin(__builtin_amdgcn_rcpf)
  return __builtin_amdgcn_rcpf(x);
#else
  return 1.0f / x;
#endif
}
static __device__ __forceinline__ float fast_sigmoid(float x){
  return fast_rcp(1.0f + __expf(-x));
}

// CDNA5 16-bit A-matrix 16x32 lane layout: lane -> M = lane%16;
// VGPR v holds K pair at: v<4 ? 2v (+8 if lane>=16) : 2(v-4)+16 (+8 if lane>=16)
static __device__ __forceinline__ int a_k_off(int v, int hi){
  return (v < 4) ? (2*v + (hi ? 8 : 0)) : (2*(v-4) + 16 + (hi ? 8 : 0));
}

static __device__ __forceinline__ v16h loadA_f32(const float* base, int stride, int kbase, int lane){
  const int m = lane & 15, hi = lane >> 4;
  HFrag f;
#pragma unroll
  for (int v = 0; v < 8; ++v){
    const int k = kbase + a_k_off(v, hi);
    f.e[2*v]   = (_Float16)base[m*stride + k];
    f.e[2*v+1] = (_Float16)base[m*stride + k + 1];
  }
  return f.h;
}

static __device__ __forceinline__ v16h loadA_f16(const _Float16* base, int stride, int kbase, int lane){
  const int m = lane & 15, hi = lane >> 4;
  const unsigned* b32 = (const unsigned*)base;  // pairs are even-aligned
  HFrag f;
#pragma unroll
  for (int v = 0; v < 8; ++v){
    const int k = kbase + a_k_off(v, hi);
    f.u[v] = b32[(m*stride + k) >> 1];
  }
  return f.h;
}

// B fragments pre-packed: frag = 32 lanes x 8 dwords (1 KB), 2x b128 per lane.
// Force addrspace(1) (via builtin vector type, trivially copyable) so these
// lower to global_load_b128 rather than flat_load_b128 (flat ticks DScnt too).
typedef __attribute__((address_space(1))) const v4u GV4u;
static __device__ __forceinline__ v16h loadB(const uint4* frags, int fragIdx, int lane){
  const GV4u* p = (const GV4u*)(uintptr_t)frags;
  HFrag f;
  f.q[0] = p[fragIdx*64 + lane*2 + 0];
  f.q[1] = p[fragIdx*64 + lane*2 + 1];
  return f.h;
}

static __device__ __forceinline__ v8f wmma16(v16h a, v16h b, v8f c){
  return __builtin_amdgcn_wmma_f32_16x16x32_f16(false, a, false, b, (short)0, c, false, false);
}

// ----- one ODE function eval: kout[16x64] = MLP(yin[16x64]) -----
static __device__ void ode_f(const float* yin, float* kout,
                             const uint4* w1f, const uint4* w2f, const uint4* w3f,
                             const float* b1, const float* b2, const float* b3,
                             _Float16* a1, _Float16* a2, int wave, int lane)
{
  const int nlo = lane & 15;
  const int m0  = (lane >> 4) * 8;

  // stage 1: a1 = tanh(yin @ W1 + b1)   [16x64]@[64x512]
  {
    AccU acc[4];
#pragma unroll
    for (int j = 0; j < 4; ++j) zero8(acc[j]);
    for (int ks = 0; ks < 2; ++ks){
      v16h A = loadA_f32(yin, L_DIM, ks*32, lane);
#pragma unroll
      for (int j = 0; j < 4; ++j){
        v16h Bf = loadB(w1f, ks*32 + wave*4 + j, lane);
        acc[j].v = wmma16(A, Bf, acc[j].v);
      }
    }
#pragma unroll
    for (int j = 0; j < 4; ++j){
      const int nt = wave*4 + j;
      const float bias = b1[nt*16 + nlo];
#pragma unroll
      for (int r = 0; r < 8; ++r)
        a1[(m0+r)*H_DIM + nt*16 + nlo] = (_Float16)fast_tanh(acc[j].f[r] + bias);
    }
  }
  __syncthreads();

  // stage 2: a2 = tanh(a1 @ W2 + b2)   [16x512]@[512x512]
  {
    AccU acc[4];
#pragma unroll
    for (int j = 0; j < 4; ++j) zero8(acc[j]);
    for (int ks = 0; ks < 16; ++ks){
      v16h A = loadA_f16(a1, H_DIM, ks*32, lane);
#pragma unroll
      for (int j = 0; j < 4; ++j){
        v16h Bf = loadB(w2f, ks*32 + wave*4 + j, lane);
        acc[j].v = wmma16(A, Bf, acc[j].v);
      }
    }
#pragma unroll
    for (int j = 0; j < 4; ++j){
      const int nt = wave*4 + j;
      const float bias = b2[nt*16 + nlo];
#pragma unroll
      for (int r = 0; r < 8; ++r)
        a2[(m0+r)*H_DIM + nt*16 + nlo] = (_Float16)fast_tanh(acc[j].f[r] + bias);
    }
  }
  __syncthreads();

  // stage 3: kout = a2 @ W3 + b3   [16x512]@[512x64]; waves 0..3
  if (wave < 4){
    AccU acc; zero8(acc);
    for (int ks = 0; ks < 16; ++ks){
      v16h A  = loadA_f16(a2, H_DIM, ks*32, lane);
      v16h Bf = loadB(w3f, ks*4 + wave, lane);
      acc.v = wmma16(A, Bf, acc.v);
    }
    const float bias = b3[wave*16 + nlo];
#pragma unroll
    for (int r = 0; r < 8; ++r)
      kout[(m0+r)*L_DIM + wave*16 + nlo] = acc.f[r] + bias;
  }
  __syncthreads();
}

// ----- RK4 over one interval with NSUB substeps -----
static __device__ void rk4_interval(float dt_total,
                                    float* ycur, float* yin, float* kout, float* kacc,
                                    const uint4* w1f, const uint4* w2f, const uint4* w3f,
                                    const float* b1, const float* b2, const float* b3,
                                    _Float16* a1, _Float16* a2,
                                    int tid, int wave, int lane)
{
  const float dt = dt_total * (1.0f / NSUB);
  for (int sub = 0; sub < NSUB; ++sub){
    for (int i = tid; i < 16*L_DIM; i += 256) yin[i] = ycur[i];
    __syncthreads();
    ode_f(yin, kout, w1f, w2f, w3f, b1, b2, b3, a1, a2, wave, lane);   // k1
    for (int i = tid; i < 16*L_DIM; i += 256){
      const float k = kout[i]; kacc[i] = k; yin[i] = ycur[i] + 0.5f*dt*k;
    }
    __syncthreads();
    ode_f(yin, kout, w1f, w2f, w3f, b1, b2, b3, a1, a2, wave, lane);   // k2
    for (int i = tid; i < 16*L_DIM; i += 256){
      const float k = kout[i]; kacc[i] += 2.f*k; yin[i] = ycur[i] + 0.5f*dt*k;
    }
    __syncthreads();
    ode_f(yin, kout, w1f, w2f, w3f, b1, b2, b3, a1, a2, wave, lane);   // k3
    for (int i = tid; i < 16*L_DIM; i += 256){
      const float k = kout[i]; kacc[i] += 2.f*k; yin[i] = ycur[i] + dt*k;
    }
    __syncthreads();
    ode_f(yin, kout, w1f, w2f, w3f, b1, b2, b3, a1, a2, wave, lane);   // k4
    for (int i = tid; i < 16*L_DIM; i += 256)
      ycur[i] += (dt * (1.0f/6.0f)) * (kacc[i] + kout[i]);
    __syncthreads();
  }
}

// ----- decoder readout: x_pred[t] = relu(z@dW1+db1)@dW2+db2 -----
static __device__ void decode_write(int t, float* out, const float* z,
                                    const uint4* dw1f, const uint4* dw2f,
                                    const float* db1, const float* db2,
                                    _Float16* a1, int b0, int wave, int lane)
{
  const int nlo = lane & 15;
  const int m0  = (lane >> 4) * 8;
  if (wave < 4){
    AccU acc; zero8(acc);
    for (int ks = 0; ks < 2; ++ks){
      v16h A  = loadA_f32(z, L_DIM, ks*32, lane);
      v16h Bf = loadB(dw1f, ks*4 + wave, lane);
      acc.v = wmma16(A, Bf, acc.v);
    }
    const float bias = db1[wave*16 + nlo];
#pragma unroll
    for (int r = 0; r < 8; ++r){
      const float v = acc.f[r] + bias;
      a1[(m0+r)*64 + wave*16 + nlo] = (_Float16)(v > 0.f ? v : 0.f);
    }
  }
  __syncthreads();
  if (wave < 2){
    AccU acc; zero8(acc);
    for (int ks = 0; ks < 2; ++ks){
      v16h A  = loadA_f16(a1, 64, ks*32, lane);
      v16h Bf = loadB(dw2f, ks*2 + wave, lane);
      acc.v = wmma16(A, Bf, acc.v);
    }
    const float bias = db2[wave*16 + nlo];
#pragma unroll
    for (int r = 0; r < 8; ++r)
      out[(size_t)t*(B_TOT*D_IN) + (size_t)(b0+m0+r)*D_IN + wave*16 + nlo] = acc.f[r] + bias;
  }
  __syncthreads();
}

// ----- persistent per-batch-tile kernel: full encoder scan + heads + decoder scan -----
__global__ void __launch_bounds__(256, 1)
latent_ode_main(const float* __restrict__ x_seq, const float* __restrict__ t_seq,
                const float* __restrict__ t_eval, const float* __restrict__ eps,
                const float* __restrict__ b1, const float* __restrict__ b2, const float* __restrict__ b3,
                const float* __restrict__ bih, const float* __restrict__ bhh,
                const float* __restrict__ mu_b, const float* __restrict__ lv_b,
                const float* __restrict__ db1, const float* __restrict__ db2,
                const uint4* __restrict__ w1f, const uint4* __restrict__ w2f, const uint4* __restrict__ w3f,
                const uint4* __restrict__ wihf, const uint4* __restrict__ whhf,
                const uint4* __restrict__ muf, const uint4* __restrict__ lvf,
                const uint4* __restrict__ dw1f, const uint4* __restrict__ dw2f,
                float* __restrict__ out)
{
  __shared__ float    s_ycur[16*L_DIM];
  __shared__ float    s_yin [16*L_DIM];
  __shared__ float    s_kout[16*L_DIM];
  __shared__ float    s_kacc[16*L_DIM];
  __shared__ _Float16 s_a1[16*H_DIM];
  __shared__ _Float16 s_a2[16*H_DIM];
  __shared__ float    s_gi[16*192];
  __shared__ float    s_gh[16*192];

  const int tid  = threadIdx.x;
  const int wave = tid >> 5;
  const int lane = tid & 31;
  const int nlo  = lane & 15;
  const int m0   = (lane >> 4) * 8;
  const int b0   = blockIdx.x * 16;

  for (int i = tid; i < 16*L_DIM; i += 256) s_ycur[i] = 0.f;   // h0 = 0
  __syncthreads();

  // ---------------- encoder scan ----------------
  for (int i = 0; i < S_LEN; ++i){
    const float dti = (i == 0) ? 0.f : (t_seq[i] - t_seq[i-1]);
    if (dti != 0.f)
      rk4_interval(dti, s_ycur, s_yin, s_kout, s_kacc,
                   w1f, w2f, w3f, b1, b2, b3, s_a1, s_a2, tid, wave, lane);

    // GRU: gi = x_i @ wih^T + bih ; gh = h @ whh^T + bhh   (N = 192)
    const float* xbase = x_seq + (size_t)b0*(S_LEN*D_IN) + (size_t)i*D_IN;
    for (int nt = wave; nt < 12; nt += 8){
      AccU acc; zero8(acc);
      v16h A  = loadA_f32(xbase, S_LEN*D_IN, 0, lane);  // K = 32 = D
      v16h Bf = loadB(wihf, nt, lane);
      acc.v = wmma16(A, Bf, acc.v);
      const float bias = bih[nt*16 + nlo];
#pragma unroll
      for (int r = 0; r < 8; ++r) s_gi[(m0+r)*192 + nt*16 + nlo] = acc.f[r] + bias;
    }
    for (int nt = wave; nt < 12; nt += 8){
      AccU acc; zero8(acc);
      for (int ks = 0; ks < 2; ++ks){
        v16h A  = loadA_f32(s_ycur, L_DIM, ks*32, lane);
        v16h Bf = loadB(whhf, ks*12 + nt, lane);
        acc.v = wmma16(A, Bf, acc.v);
      }
      const float bias = bhh[nt*16 + nlo];
#pragma unroll
      for (int r = 0; r < 8; ++r) s_gh[(m0+r)*192 + nt*16 + nlo] = acc.f[r] + bias;
    }
    __syncthreads();
    for (int idx = tid; idx < 16*L_DIM; idx += 256){
      const int m = idx >> 6, j = idx & 63;
      const float r = fast_sigmoid(s_gi[m*192 + j]       + s_gh[m*192 + j]);
      const float z = fast_sigmoid(s_gi[m*192 + 64 + j]  + s_gh[m*192 + 64 + j]);
      const float n = fast_tanh   (s_gi[m*192 + 128 + j] + r * s_gh[m*192 + 128 + j]);
      s_ycur[idx] = (1.f - z) * n + z * s_ycur[idx];
    }
    __syncthreads();
  }

  // ---------------- mu / logvar heads + reparameterize ----------------
  {
    const int sel = wave >> 2;   // 0: mu, 1: logvar
    const int nt  = wave & 3;
    AccU acc; zero8(acc);
    for (int ks = 0; ks < 2; ++ks){
      v16h A  = loadA_f32(s_ycur, L_DIM, ks*32, lane);
      v16h Bf = loadB(sel ? lvf : muf, ks*4 + nt, lane);
      acc.v = wmma16(A, Bf, acc.v);
    }
    const float bias = (sel ? lv_b : mu_b)[nt*16 + nlo];
#pragma unroll
    for (int r = 0; r < 8; ++r){
      const float v = acc.f[r] + bias;
      s_gi[sel*1024 + (m0+r)*64 + nt*16 + nlo] = v;
      out[(size_t)T_EVAL*B_TOT*D_IN + (size_t)sel*(B_TOT*L_DIM)
          + (size_t)(b0+m0+r)*L_DIM + nt*16 + nlo] = v;
    }
  }
  __syncthreads();
  for (int idx = tid; idx < 16*L_DIM; idx += 256){
    const int m = idx >> 6, j = idx & 63;
    s_ycur[idx] = s_gi[idx] + eps[(size_t)(b0+m)*L_DIM + j] * __expf(0.5f*s_gi[1024 + idx]);
  }
  __syncthreads();

  // ---------------- decoder scan ----------------
  decode_write(0, out, s_ycur, dw1f, dw2f, db1, db2, s_a1, b0, wave, lane);
  for (int t = 1; t < T_EVAL; ++t){
    const float dtt = t_eval[t] - t_eval[t-1];
    rk4_interval(dtt, s_ycur, s_yin, s_kout, s_kacc,
                 w1f, w2f, w3f, b1, b2, b3, s_a1, s_a2, tid, wave, lane);
    decode_write(t, out, s_ycur, dw1f, dw2f, db1, db2, s_a1, b0, wave, lane);
  }
}

// ----- weight packer: f32 [K][N] (or transposed [N][K]) -> f16 B-fragment-major -----
__global__ void pack_b_kernel(const float* __restrict__ src, unsigned* __restrict__ dst,
                              int ksteps, int ntiles, int N, int K, int transpose)
{
  const int tid   = blockIdx.x * blockDim.x + threadIdx.x;
  const int total = ksteps * ntiles * 32;
  if (tid >= total) return;
  const int lane = tid & 31;
  const int frag = tid >> 5;
  const int ks = frag / ntiles, nt = frag % ntiles;
  const int n  = nt*16 + (lane & 15);
  const int kb = ks*32 + ((lane >= 16) ? 8 : 0);
  unsigned* o = dst + (size_t)frag*256 + lane*8;
#pragma unroll
  for (int v = 0; v < 8; ++v){
    const int k = kb + ((v < 4) ? 2*v : 2*(v-4) + 16);
    const float x0 = transpose ? src[(size_t)n*K + k]     : src[(size_t)k*N + n];
    const float x1 = transpose ? src[(size_t)n*K + k + 1] : src[(size_t)(k+1)*N + n];
    Pack2 p; p.e[0] = (_Float16)x0; p.e[1] = (_Float16)x1;
    o[v] = p.u;
  }
  (void)N; (void)K;
}

extern "C" void kernel_launch(void* const* d_in, const int* in_sizes, int n_in,
                              void* d_out, int out_size, void* d_ws, size_t ws_size,
                              hipStream_t stream) {
  (void)in_sizes; (void)n_in; (void)out_size; (void)ws_size;
  const float* x_seq  = (const float*)d_in[0];
  const float* t_seq  = (const float*)d_in[1];
  const float* t_ev   = (const float*)d_in[2];
  const float* eps    = (const float*)d_in[3];
  const float* ode_w1 = (const float*)d_in[4];
  const float* ode_b1 = (const float*)d_in[5];
  const float* ode_w2 = (const float*)d_in[6];
  const float* ode_b2 = (const float*)d_in[7];
  const float* ode_w3 = (const float*)d_in[8];
  const float* ode_b3 = (const float*)d_in[9];
  const float* gru_wih = (const float*)d_in[10];
  const float* gru_whh = (const float*)d_in[11];
  const float* gru_bih = (const float*)d_in[12];
  const float* gru_bhh = (const float*)d_in[13];
  const float* mu_w = (const float*)d_in[14];
  const float* mu_b = (const float*)d_in[15];
  const float* lv_w = (const float*)d_in[16];
  const float* lv_b = (const float*)d_in[17];
  const float* dec_w1 = (const float*)d_in[18];
  const float* dec_b1 = (const float*)d_in[19];
  const float* dec_w2 = (const float*)d_in[20];
  const float* dec_b2 = (const float*)d_in[21];
  float* out = (float*)d_out;

  unsigned* ws = (unsigned*)d_ws;
  size_t off = 0;
  auto alloc = [&](int ksteps, int ntiles){
    unsigned* p = ws + off; off += (size_t)ksteps * ntiles * 256; return p;
  };
  unsigned* w1f  = alloc(2, 32);
  unsigned* w2f  = alloc(16, 32);
  unsigned* w3f  = alloc(16, 4);
  unsigned* wihf = alloc(1, 12);
  unsigned* whhf = alloc(2, 12);
  unsigned* muf  = alloc(2, 4);
  unsigned* lvf  = alloc(2, 4);
  unsigned* dw1f = alloc(2, 4);
  unsigned* dw2f = alloc(2, 2);

  auto pack = [&](const float* s, unsigned* d, int ksteps, int ntiles, int N, int K, int tr){
    const int total = ksteps * ntiles * 32;
    pack_b_kernel<<<(total + 255) / 256, 256, 0, stream>>>(s, d, ksteps, ntiles, N, K, tr);
  };
  pack(ode_w1, w1f, 2, 32, 512, 64, 0);    // [64,512]
  pack(ode_w2, w2f, 16, 32, 512, 512, 0);  // [512,512]
  pack(ode_w3, w3f, 16, 4, 64, 512, 0);    // [512,64]
  pack(gru_wih, wihf, 1, 12, 192, 32, 1);  // B = wih^T, [32,192]
  pack(gru_whh, whhf, 2, 12, 192, 64, 1);  // B = whh^T, [64,192]
  pack(mu_w, muf, 2, 4, 64, 64, 0);
  pack(lv_w, lvf, 2, 4, 64, 64, 0);
  pack(dec_w1, dw1f, 2, 4, 64, 64, 0);
  pack(dec_w2, dw2f, 2, 2, 32, 64, 0);

  latent_ode_main<<<B_TOT / 16, 256, 0, stream>>>(
      x_seq, t_seq, t_ev, eps,
      ode_b1, ode_b2, ode_b3, gru_bih, gru_bhh, mu_b, lv_b, dec_b1, dec_b2,
      (const uint4*)w1f, (const uint4*)w2f, (const uint4*)w3f,
      (const uint4*)wihf, (const uint4*)whhf,
      (const uint4*)muf, (const uint4*)lvf,
      (const uint4*)dw1f, (const uint4*)dw2f,
      out);
}
